// SparseSAKELayer_20564303413686
// MI455X (gfx1250) — matrix operand order repaired and checked by reading, not compiled, not run on hardware
//
#include <hip/hip_runtime.h>

// ---------------------------------------------------------------------------
// SAKE layer for MI455X (gfx1250): bf16 WMMA for edge GEMMs, fp32 accumulate.
// Weights pre-swizzled into WMMA B-fragment order; A operands staged in LDS
// as bf16 so fragment builds are pure b128 loads (no per-step converts).
// ---------------------------------------------------------------------------

typedef __attribute__((ext_vector_type(16))) __bf16 v16bf;
typedef __attribute__((ext_vector_type(8)))  __bf16 v8bf;
typedef __attribute__((ext_vector_type(8)))  float  v8f;

#define EPB 16   // edges per block for edge kernels

__device__ inline float siluf(float x)    { return x / (1.0f + __expf(-x)); }
__device__ inline float sigmoidf_(float x){ return 1.0f / (1.0f + __expf(-x)); }

__device__ inline __bf16 f2bf(float f) {
    unsigned u = __float_as_uint(f);
    u += 0x7FFFu + ((u >> 16) & 1u);          // round-to-nearest-even
    unsigned short s = (unsigned short)(u >> 16);
    __bf16 r;
    __builtin_memcpy(&r, &s, sizeof(r));
    return r;
}

// A fragment from a bf16 LDS row. ISA 7.12.2 16-bit A 16x32: lane L holds row
// M=L&15; elems 0..7 -> K = k0+(L>>4)*8+i ; elems 8..15 -> K = k0+16+(L>>4)*8+i.
// Both runs are 8 contiguous bf16 = 16 bytes -> two ds_load_b128.
__device__ inline v16bf make_a(const __bf16* row, int k0, int hi) {
    union { v16bf v; v8bf h[2]; } u;
    u.h[0] = *(const v8bf*)(row + k0 + hi * 8);
    u.h[1] = *(const v8bf*)(row + k0 + 16 + hi * 8);
    return u.v;
}

// B fragment from pre-swizzled weights: fragment (tile,kk) holds 32 lanes x
// 16 bf16 contiguously -> one 32-byte load per lane.
__device__ inline v16bf make_b(const __bf16* Wf, int frag, int lane) {
    return *(const v16bf*)(Wf + ((size_t)frag * 32 + lane) * 16);
}

__device__ inline v8f wmma_bf16(v16bf a, v16bf b, v8f c) {
    return __builtin_amdgcn_wmma_f32_16x16x32_bf16(false, a, false, b, (short)0, c,
                                                   false, false);
}

__device__ inline unsigned float_order_key(float f) {
    unsigned u = __float_as_uint(f);
    return (u & 0x80000000u) ? ~u : (u | 0x80000000u);
}
__device__ inline float float_order_dec(unsigned v) {
    return (v & 0x80000000u) ? __uint_as_float(v ^ 0x80000000u) : __uint_as_float(~v);
}

// ---------------------------------------------------------------------------
// Weight conversion to bf16 in swizzled WMMA B-fragment layout.
// For a matrix with KS k-steps and NT n-tiles (Ntot = NT*16, Ktot = KS*32):
//   frag = tile*KS + kk ; element (frag, lane, i) <- W[k][n] with
//   n = tile*16 + (lane&15), k = kk*32 + (i>>3)*16 + (lane>>4)*8 + (i&7)  (zero padded OOB)
// win: K=128,N=64(src 128x50) | wout1: K=192(src 179),N=64 | wout2: 64x64
// wx: 256x256
// ---------------------------------------------------------------------------
__device__ inline void swizzle_one(const float* W, int ldn, int Ksrc, int Nsrc,
                                   int KS, __bf16* dst, int idx) {
    int i    = idx & 15;
    int lane = (idx >> 4) & 31;
    int frag = idx >> 9;
    int tile = frag / KS, kk = frag % KS;
    int n = tile * 16 + (lane & 15);
    int k = kk * 32 + (i >> 3) * 16 + (lane >> 4) * 8 + (i & 7);
    float v = (k < Ksrc && n < Nsrc) ? W[k * ldn + n] : 0.0f;
    dst[idx] = f2bf(v);
}

__global__ void k_wconv(const float* __restrict__ W_in, const float* __restrict__ W_out1,
                        const float* __restrict__ W_out2, const float* __restrict__ W_xmix,
                        __bf16* win, __bf16* wout1, __bf16* wout2, __bf16* wx) {
    const int n_win = 4 * 4 * 512;    // NT=4, KS=4
    const int n_w1  = 4 * 6 * 512;    // NT=4, KS=6
    const int n_w2  = 4 * 2 * 512;    // NT=4, KS=2
    const int n_wx  = 16 * 8 * 512;   // NT=16, KS=8
    const int total = n_win + n_w1 + n_w2 + n_wx;
    for (int idx = blockIdx.x * blockDim.x + threadIdx.x; idx < total;
         idx += gridDim.x * blockDim.x) {
        int i = idx;
        if (i < n_win) { swizzle_one(W_in, 50, 128, 50, 4, win, i); continue; }
        i -= n_win;
        if (i < n_w1)  { swizzle_one(W_out1, 64, 179, 64, 6, wout1, i); continue; }
        i -= n_w1;
        if (i < n_w2)  { swizzle_one(W_out2, 64, 64, 64, 2, wout2, i); continue; }
        i -= n_w2;
        swizzle_one(W_xmix, 256, 256, 256, 8, wx, i);
    }
}

// ---------------------------------------------------------------------------
// Edge kernel 1: gather + RBF + (h_in, mid, h_e_mtx) WMMA chain + att logits
// ---------------------------------------------------------------------------
__global__ __launch_bounds__(128) void k_edge1(
    const float* __restrict__ h, const float* __restrict__ x, const int* __restrict__ idxs,
    const float* __restrict__ means, const float* __restrict__ betas,
    const float* __restrict__ b_in, const float* __restrict__ b_out1,
    const float* __restrict__ b_out2, const float* __restrict__ W_att,
    const float* __restrict__ b_att,
    const __bf16* __restrict__ win, const __bf16* __restrict__ wout1,
    const __bf16* __restrict__ wout2,
    float* __restrict__ he_mtx_g, float* __restrict__ xhat_g, float* __restrict__ att_g,
    unsigned* __restrict__ segmax, float* __restrict__ counts, int E_) {
    __shared__ alignas(32) __bf16 feat[EPB][192]; // [h_cat(128)|_x(50)|xn(1)|pad(13)]
    __shared__ alignas(32) __bf16 mid_s[EPB][64];
    __shared__ float he_s[EPB][64];
    __shared__ float xn_s[EPB];
    __shared__ int   src_s[EPB];
    __shared__ int   dst_s[EPB];

    const int tid = threadIdx.x;
    const int e0  = blockIdx.x * EPB;

    if (tid < EPB) {
        int e = e0 + tid;
        if (e >= E_) e = E_ - 1;   // E is a multiple of 16; clamp only for safety
        int s = idxs[2 * e], d = idxs[2 * e + 1];
        src_s[tid] = s; dst_s[tid] = d;
        float dx = x[3 * s + 0] - x[3 * d + 0];
        float dy = x[3 * s + 1] - x[3 * d + 1];
        float dz = x[3 * s + 2] - x[3 * d + 2];
        float xn = sqrtf(dx * dx + dy * dy + dz * dz + 1e-14f);
        xn_s[tid] = xn;
        float inv = 1.0f / (xn + 1e-5f);
        xhat_g[3 * e + 0] = dx * inv;
        xhat_g[3 * e + 1] = dy * inv;
        xhat_g[3 * e + 2] = dz * inv;
        atomicAdd(&counts[s], 1.0f);
        feat[tid][178] = f2bf(xn);
#pragma unroll
        for (int c = 179; c < 192; ++c) feat[tid][c] = f2bf(0.0f);
    }
    __syncthreads();
    for (int i = tid; i < EPB * 128; i += 128) {
        int m = i >> 7, c = i & 127;
        float v = (c < 64) ? h[src_s[m] * 64 + c] : h[dst_s[m] * 64 + (c - 64)];
        feat[m][c] = f2bf(v);
    }
    __syncthreads();

    const int lane = tid & 31, wv = tid >> 5;
    const int hi = lane >> 4, ln = lane & 15;

    // ---- h_in = h_cat @ W_in (K=128, KS=4); then _x = rbf*h_in back to feat ----
    {
        v8f acc; for (int r = 0; r < 8; ++r) acc[r] = 0.0f;
#pragma unroll
        for (int kk = 0; kk < 4; ++kk) {
            v16bf a = make_a(&feat[ln][0], kk * 32, hi);
            v16bf b = make_b(win, wv * 4 + kk, lane);
            acc = wmma_bf16(a, b, acc);
        }
        int n = wv * 16 + ln;
        if (n < 50) {
            float mean = means[n], beta = betas[n], bi = b_in[n];
#pragma unroll
            for (int r = 0; r < 8; ++r) {
                int m = r + hi * 8;
                float hin = acc[r] + bi;
                float t = __expf(-xn_s[m]) - mean;   // ALPHA = 1.0, CUTOFF_LOWER = 0
                float rbf = __expf(-beta * t * t);
                feat[m][128 + n] = f2bf(rbf * hin);
            }
        }
    }
    __syncthreads();

    // ---- mid = silu(feat @ W_out1 + b_out1), K=192, KS=6 ----
    {
        v8f acc; for (int r = 0; r < 8; ++r) acc[r] = 0.0f;
#pragma unroll
        for (int kk = 0; kk < 6; ++kk) {
            v16bf a = make_a(&feat[ln][0], kk * 32, hi);
            v16bf b = make_b(wout1, wv * 6 + kk, lane);
            acc = wmma_bf16(a, b, acc);
        }
        int n = wv * 16 + ln;
        float bo = b_out1[n];
#pragma unroll
        for (int r = 0; r < 8; ++r) {
            int m = r + hi * 8;
            mid_s[m][n] = f2bf(siluf(acc[r] + bo));
        }
    }
    __syncthreads();

    // ---- h_e_mtx = mid @ W_out2 + b_out2, K=64, KS=2 ----
    {
        v8f acc; for (int r = 0; r < 8; ++r) acc[r] = 0.0f;
#pragma unroll
        for (int kk = 0; kk < 2; ++kk) {
            v16bf a = make_a(&mid_s[ln][0], kk * 32, hi);
            v16bf b = make_b(wout2, wv * 2 + kk, lane);
            acc = wmma_bf16(a, b, acc);
        }
        int n = wv * 16 + ln;
        float bo = b_out2[n];
#pragma unroll
        for (int r = 0; r < 8; ++r) {
            int m = r + hi * 8;
            float v = acc[r] + bo;
            he_s[m][n] = v;
            he_mtx_g[(long)(e0 + m) * 64 + n] = v;
        }
    }
    __syncthreads();

    // ---- att logits = celu(h_e_mtx @ W_att + b_att, alpha=2) + segment max ----
    if (tid < 64) {
        int m = tid >> 2, a = tid & 3;
        float s = b_att[a];
#pragma unroll 8
        for (int j = 0; j < 64; ++j) s += he_s[m][j] * W_att[j * 4 + a];
        float c = (s > 0.0f) ? s : 2.0f * (__expf(s * 0.5f) - 1.0f);
        att_g[(long)(e0 + m) * 4 + a] = c;
        atomicMax(&segmax[src_s[m] * 4 + a], float_order_key(c));
    }
}

// ---------------------------------------------------------------------------
// Softmax pass: att := exp(att - segmax[src]); segsum += att
// (comb_att = att/segsum then /segment_sum(att)==1 -> softmax itself)
// ---------------------------------------------------------------------------
__global__ void k_soft(const int* __restrict__ idxs, float* __restrict__ att_g,
                       const unsigned* __restrict__ segmax, float* __restrict__ segsum,
                       int E_) {
    int i = blockIdx.x * blockDim.x + threadIdx.x;
    if (i >= E_ * 4) return;
    int e = i >> 2, a = i & 3;
    int s = idxs[2 * e];
    float m = float_order_dec(segmax[s * 4 + a]);
    float ex = __expf(att_g[i] - m);
    att_g[i] = ex;
    atomicAdd(&segsum[s * 4 + a], ex);
}

// ---------------------------------------------------------------------------
// Edge kernel 2: h_e_att build + h_e scatter + coeff = tanh(h_e_att @ W_xmix)
// + scatter combinations = coeff (x) xhat into comb_raw
// ---------------------------------------------------------------------------
__global__ __launch_bounds__(128) void k_edge2(
    const int* __restrict__ idxs, const float* __restrict__ he_mtx_g,
    const float* __restrict__ att_g, const float* __restrict__ segsum,
    const float* __restrict__ xhat_g, const __bf16* __restrict__ wx,
    float* __restrict__ h_e_acc, float* __restrict__ comb_raw, int E_) {
    __shared__ alignas(32) __bf16 heatt[EPB][256];  // 8 KB
    __shared__ float he_s[EPB][64];                 // staged h_e_mtx tile
    __shared__ float ca_s[EPB][4];
    __shared__ float xh_s[EPB][3];
    __shared__ int   src_s[EPB];

    const int tid = threadIdx.x;
    const int e0  = blockIdx.x * EPB;

    if (tid < EPB) {
        int e = e0 + tid;
        if (e >= E_) e = E_ - 1;
        src_s[tid] = idxs[2 * e];
        xh_s[tid][0] = xhat_g[3 * e + 0];
        xh_s[tid][1] = xhat_g[3 * e + 1];
        xh_s[tid][2] = xhat_g[3 * e + 2];
    }
    __syncthreads();
    if (tid < 64) {
        int m = tid >> 2, a = tid & 3;
        ca_s[m][a] = att_g[(long)(e0 + m) * 4 + a] / segsum[src_s[m] * 4 + a];
    }
    for (int i = tid; i < EPB * 64; i += 128) {
        int m = i >> 6, f = i & 63;
        he_s[m][f] = he_mtx_g[(long)(e0 + m) * 64 + f];
    }
    __syncthreads();
    for (int i = tid; i < EPB * 256; i += 128) {
        int m = i >> 8, c = i & 255;
        int f = c >> 2, a = c & 3;   // h_e_att[e, f*4+a] = h_e_mtx[e,f]*comb_att[e,a]
        float v = he_s[m][f] * ca_s[m][a];
        heatt[m][c] = f2bf(v);
        atomicAdd(&h_e_acc[(long)src_s[m] * 256 + c], v);
    }
    __syncthreads();

    const int lane = tid & 31, wv = tid >> 5;
    const int hi = lane >> 4, ln = lane & 15;
    for (int t = 0; t < 4; ++t) {
        const int tile = wv * 4 + t;
        v8f acc; for (int r = 0; r < 8; ++r) acc[r] = 0.0f;
#pragma unroll
        for (int kk = 0; kk < 8; ++kk) {
            v16bf a = make_a(&heatt[ln][0], kk * 32, hi);
            v16bf b = make_b(wx, tile * 8 + kk, lane);
            acc = wmma_bf16(a, b, acc);
        }
#pragma unroll
        for (int r = 0; r < 8; ++r) {
            int m = r + hi * 8;
            int c = tile * 16 + ln;
            float co = tanhf(acc[r]);
            float* dst = &comb_raw[(long)src_s[m] * 768 + c * 3];
            atomicAdd(dst + 0, co * xh_s[m][0]);
            atomicAdd(dst + 1, co * xh_s[m][1]);
            atomicAdd(dst + 2, co * xh_s[m][2]);
        }
    }
}

// ---------------------------------------------------------------------------
// Node kernel: one wave per node; small MLPs in fp32 VALU; writes d_out
// d_out layout: h_new[N*64] | x_new[N*3] | v_new[N*3]
// ---------------------------------------------------------------------------
__global__ __launch_bounds__(128) void k_node(
    const float* __restrict__ h, const float* __restrict__ x, const float* __restrict__ v,
    const float* __restrict__ counts, const float* __restrict__ comb_raw,
    const float* __restrict__ h_e_acc,
    const float* __restrict__ W_post1, const float* __restrict__ b_post1,
    const float* __restrict__ W_post2, const float* __restrict__ b_post2,
    const float* __restrict__ W_node1, const float* __restrict__ b_node1,
    const float* __restrict__ W_node2, const float* __restrict__ b_node2,
    const float* __restrict__ W_vmix, const float* __restrict__ W_vel1,
    const float* __restrict__ b_vel1, const float* __restrict__ W_vel2,
    float* __restrict__ out, int N_) {
    __shared__ float cs[4][768];    // comb_sum (divided)
    __shared__ float cn[4][256];    // comb_norm
    __shared__ float nin[4][384];   // [h | h_e | h_comb]
    __shared__ float t1[4][64];
    __shared__ float t2[4][64];
    __shared__ float hn[4][64];
    __shared__ float sc_s[4];

    const int tid = threadIdx.x, lane = tid & 31, wv = tid >> 5;
    const int n = blockIdx.x * 4 + wv;
    const bool valid = (n < N_);

    if (valid) {
        float inv = 1.0f / (counts[n] + 1.0f);
        const float* cr = &comb_raw[(long)n * 768];
        for (int i = lane; i < 768; i += 32) cs[wv][i] = cr[i] * inv;
    }
    __syncthreads();
    if (valid) {
        for (int i = lane; i < 256; i += 32) {
            float a = cs[wv][3 * i], b = cs[wv][3 * i + 1], c = cs[wv][3 * i + 2];
            cn[wv][i] = a * a + b * b + c * c;
        }
        for (int i = lane; i < 64; i += 32)  nin[wv][i] = h[(long)n * 64 + i];
        for (int i = lane; i < 256; i += 32) nin[wv][64 + i] = h_e_acc[(long)n * 256 + i];
    }
    __syncthreads();
    if (valid) {  // h_comb layer 1
        for (int j = lane; j < 64; j += 32) {
            float s = b_post1[j];
            for (int k = 0; k < 256; ++k) s += cn[wv][k] * W_post1[k * 64 + j];
            t1[wv][j] = siluf(s);
        }
    }
    __syncthreads();
    if (valid) {  // h_comb layer 2 -> nin[320..383]
        for (int j = lane; j < 64; j += 32) {
            float s = b_post2[j];
            for (int k = 0; k < 64; ++k) s += t1[wv][k] * W_post2[k * 64 + j];
            nin[wv][320 + j] = siluf(s);
        }
    }
    __syncthreads();
    if (valid) {  // node MLP layer 1
        for (int j = lane; j < 64; j += 32) {
            float s = b_node1[j];
            for (int k = 0; k < 384; ++k) s += nin[wv][k] * W_node1[k * 64 + j];
            t2[wv][j] = siluf(s);
        }
    }
    __syncthreads();
    if (valid) {  // node MLP layer 2 + residual -> h_new
        for (int j = lane; j < 64; j += 32) {
            float s = b_node2[j];
            for (int k = 0; k < 64; ++k) s += t2[wv][k] * W_node2[k * 64 + j];
            float hv = h[(long)n * 64 + j] + siluf(s);
            hn[wv][j] = hv;
            out[(long)n * 64 + j] = hv;
        }
    }
    __syncthreads();
    if (valid) {  // velocity gate layer 1 (fold W_vel2 into partials)
        for (int j = lane; j < 64; j += 32) {
            float s = b_vel1[j];
            for (int k = 0; k < 64; ++k) s += hn[wv][k] * W_vel1[k * 64 + j];
            t1[wv][j] = siluf(s) * W_vel2[j];
        }
    }
    __syncthreads();
    if (valid && lane == 0) {
        float s = 0.0f;
        for (int k = 0; k < 64; ++k) s += t1[wv][k];
        sc_s[wv] = 2.0f * sigmoidf_(s);
    }
    __syncthreads();
    if (valid && lane < 3) {
        int d = lane;
        float s = 0.0f;   // dv[d] = sum_c comb_sum[c][d]*W_vmix[c]
        for (int c = 0; c < 256; ++c) s += cs[wv][3 * c + d] * W_vmix[c];
        float vn = s + sc_s[wv] * v[(long)n * 3 + d];
        float xn = x[(long)n * 3 + d] + vn;
        long base = (long)N_ * 64;
        out[base + (long)n * 3 + d] = xn;
        out[base + (long)N_ * 3 + (long)n * 3 + d] = vn;
    }
}

// ---------------------------------------------------------------------------
extern "C" void kernel_launch(void* const* d_in, const int* in_sizes, int n_in,
                              void* d_out, int out_size, void* d_ws, size_t ws_size,
                              hipStream_t stream) {
    (void)n_in; (void)out_size; (void)ws_size;
    const float* h       = (const float*)d_in[0];
    const float* x       = (const float*)d_in[1];
    const float* v       = (const float*)d_in[2];
    const int*   idxs    = (const int*)d_in[3];
    const float* means   = (const float*)d_in[4];
    const float* betas   = (const float*)d_in[5];
    const float* W_in    = (const float*)d_in[6];
    const float* b_in    = (const float*)d_in[7];
    const float* W_out1  = (const float*)d_in[8];
    const float* b_out1  = (const float*)d_in[9];
    const float* W_out2  = (const float*)d_in[10];
    const float* b_out2  = (const float*)d_in[11];
    const float* W_att   = (const float*)d_in[12];
    const float* b_att   = (const float*)d_in[13];
    const float* W_xmix  = (const float*)d_in[14];
    const float* W_post1 = (const float*)d_in[15];
    const float* b_post1 = (const float*)d_in[16];
    const float* W_post2 = (const float*)d_in[17];
    const float* b_post2 = (const float*)d_in[18];
    const float* W_node1 = (const float*)d_in[19];
    const float* b_node1 = (const float*)d_in[20];
    const float* W_node2 = (const float*)d_in[21];
    const float* b_node2 = (const float*)d_in[22];
    const float* W_vmix  = (const float*)d_in[23];
    const float* W_vel1  = (const float*)d_in[24];
    const float* b_vel1  = (const float*)d_in[25];
    const float* W_vel2  = (const float*)d_in[26];

    const int N_ = in_sizes[0] / 64;  // 10000
    const int E_ = in_sizes[3] / 2;   // 160000

    char* ws = (char*)d_ws;
    size_t off = 0;
    auto alloc = [&](size_t bytes) -> void* {
        void* p = ws + off;
        off += (bytes + 255) & ~(size_t)255;
        return p;
    };
    // zeroed accumulator region (contiguous at front)
    float*    comb_raw = (float*)alloc((size_t)N_ * 768 * 4);
    float*    h_e_acc  = (float*)alloc((size_t)N_ * 256 * 4);
    float*    segsum   = (float*)alloc((size_t)N_ * 4 * 4);
    unsigned* segmax   = (unsigned*)alloc((size_t)N_ * 4 * 4);
    float*    counts   = (float*)alloc((size_t)N_ * 4);
    const size_t zero_bytes = off;
    // non-zeroed scratch
    float*  he_mtx = (float*)alloc((size_t)E_ * 64 * 4);
    float*  xhat   = (float*)alloc((size_t)E_ * 3 * 4);
    float*  att    = (float*)alloc((size_t)E_ * 4 * 4);
    __bf16* win    = (__bf16*)alloc((size_t)4 * 4 * 512 * 2);
    __bf16* wout1  = (__bf16*)alloc((size_t)4 * 6 * 512 * 2);
    __bf16* wout2  = (__bf16*)alloc((size_t)4 * 2 * 512 * 2);
    __bf16* wx     = (__bf16*)alloc((size_t)16 * 8 * 512 * 2);

    hipMemsetAsync(d_ws, 0, zero_bytes, stream);

    k_wconv<<<352, 256, 0, stream>>>(W_in, W_out1, W_out2, W_xmix, win, wout1, wout2, wx);

    const int eb = (E_ + EPB - 1) / EPB;
    k_edge1<<<eb, 128, 0, stream>>>(h, x, idxs, means, betas, b_in, b_out1, b_out2,
                                    W_att, b_att, win, wout1, wout2,
                                    he_mtx, xhat, att, segmax, counts, E_);

    k_soft<<<(E_ * 4 + 255) / 256, 256, 0, stream>>>(idxs, att, segmax, segsum, E_);

    k_edge2<<<eb, 128, 0, stream>>>(idxs, he_mtx, att, segsum, xhat, wx,
                                    h_e_acc, comb_raw, E_);

    k_node<<<(N_ + 3) / 4, 128, 0, stream>>>(h, x, v, counts, comb_raw, h_e_acc,
                                             W_post1, b_post1, W_post2, b_post2,
                                             W_node1, b_node1, W_node2, b_node2,
                                             W_vmix, W_vel1, b_vel1, W_vel2,
                                             (float*)d_out, N_);
}
